// ConvLayer_48430051230519
// MI455X (gfx1250) — matrix operand order repaired and checked by reading, not compiled, not run on hardware
//
#include <hip/hip_runtime.h>

typedef __attribute__((ext_vector_type(16))) _Float16 v16h;
typedef __attribute__((ext_vector_type(8)))  _Float16 v8h;
typedef __attribute__((ext_vector_type(8)))  float    v8f;

// ---------------------------------------------------------------------------
// WMMA fragment helpers (layouts per CDNA5 ISA 7.12.2, wave32)
// ---------------------------------------------------------------------------

// A-matrix 16x32 f16 fragment from row-major LDS tile (ldk halfs per row).
// lane 0-15: row=lane, K = kk+{0..7, 16..23}; lane 16-31: K = kk+{8..15, 24..31}
__device__ __forceinline__ v16h ldA_frag(const _Float16* A, int ldk, int lane, int kk) {
  int row = lane & 15;
  int h   = lane >> 4;
  const _Float16* p = A + row * ldk + kk + h * 8;
  v8h lo = *(const v8h*)(p);        // K = kk + h*8 + 0..7
  v8h hi = *(const v8h*)(p + 16);   // K = kk + 16 + h*8 + 0..7
  return __builtin_shufflevector(lo, hi, 0,1,2,3,4,5,6,7,8,9,10,11,12,13,14,15);
}

// B-matrix 32x16 f16 fragment from K-major ("transposed", [Ncol][K]) weights:
// lane holds K = {8h..8h+7, 16+8h..16+8h+7} of one column -> two 16B loads.
__device__ __forceinline__ v16h ldB_frag(const _Float16* p) {
  const _Float16* q = p - ((threadIdx.x & 31) >> 4) * 8;
  v8h lo = *(const v8h*)(q);
  v8h hi = *(const v8h*)(q + 16);
  return __builtin_shufflevector(lo, hi, 0,1,2,3,4,5,6,7,8,9,10,11,12,13,14,15);
}

__device__ __forceinline__ v8f wmma_f16(v16h a, v16h b, v8f c) {
  // 8 args: (neg_a, A, neg_b, B, c_mod, C, reuse_a, reuse_b)
  return __builtin_amdgcn_wmma_f32_16x16x32_f16(false, a, false, b, (short)0, c, false, false);
}

__device__ __forceinline__ v8f vzero8() {
  v8f c = {0.f,0.f,0.f,0.f,0.f,0.f,0.f,0.f};
  return c;
}

// ---------------------------------------------------------------------------
// Kernel 1: transpose+pad weights to f16 [N][K] layout, zero BN stat buffers
// ---------------------------------------------------------------------------
__global__ void k_prep(const float* __restrict__ W1, const float* __restrict__ W2,
                       const float* __restrict__ W3, const float* __restrict__ eW1,
                       const float* __restrict__ eW2,
                       _Float16* __restrict__ W1t, _Float16* __restrict__ W2t,
                       _Float16* __restrict__ W3t, _Float16* __restrict__ eW1t,
                       _Float16* __restrict__ eW2t,
                       float* __restrict__ stats_in, float* __restrict__ stats_out) {
  int i = blockIdx.x * 256 + threadIdx.x;
  if (i < 16384) {                       // W1t[256][64], K padded 41->64
    int n = i >> 6, k = i & 63;
    W1t[i] = (k < 41) ? (_Float16)W1[k * 256 + n] : (_Float16)0.f;
  } else if (i < 49152) {                // W2t[128][256]
    int j = i - 16384; int n = j >> 8, k = j & 255;
    W2t[j] = (_Float16)W2[k * 128 + n];
  } else if (i < 57344) {                // W3t[64][128]
    int j = i - 49152; int n = j >> 7, k = j & 127;
    W3t[j] = (_Float16)W3[k * 64 + n];
  } else if (i < 90112) {                // eW1t[256][128]
    int j = i - 57344; int n = j >> 7, k = j & 127;
    eW1t[j] = (_Float16)eW1[k * 256 + n];
  } else if (i < 106496) {               // eW2t[64][256]
    int j = i - 90112; int n = j >> 8, k = j & 255;
    eW2t[j] = (_Float16)eW2[k * 64 + n];
  } else if (i < 106752) {
    int j = i - 106496;
    if (j < 128) stats_in[j] = 0.f; else stats_out[j - 128] = 0.f;
  }
}

// ---------------------------------------------------------------------------
// Kernel 2/6: per-feature sum & sumsq over N rows (64 features)
// ---------------------------------------------------------------------------
__global__ void k_stats(const float* __restrict__ X, const float* __restrict__ cnt,
                        float* __restrict__ stats, int n) {
  int f = threadIdx.x & 63;
  int g = threadIdx.x >> 6;              // 4 row-groups per block
  float s = 0.f, s2 = 0.f;
  for (int r = blockIdx.x * 4 + g; r < n; r += gridDim.x * 4) {
    float v = X[r * 64 + f];
    if (cnt) v /= fmaxf(cnt[r], 1.f);
    s += v; s2 += v * v;
  }
  atomicAdd(&stats[f], s);
  atomicAdd(&stats[64 + f], s2);
}

// ---------------------------------------------------------------------------
// Kernel 3: x = BN(atom_fea); combined = x; agg = 0; cnt = 0
// ---------------------------------------------------------------------------
__global__ void k_apply_in(const float* __restrict__ atom, const float* __restrict__ stats,
                           const float* __restrict__ g, const float* __restrict__ b,
                           float* __restrict__ x, float* __restrict__ combined,
                           float* __restrict__ agg, float* __restrict__ cnt, int n) {
  int i = blockIdx.x * 256 + threadIdx.x;
  if (i >= n * 64) return;
  int f = i & 63;
  float inv_n = 1.f / (float)n;
  float mu  = stats[f] * inv_n;
  float var = stats[64 + f] * inv_n - mu * mu;
  float xv = (atom[i] - mu) * rsqrtf(var + 1e-5f) * g[f] + b[f];
  x[i] = xv;
  combined[i] = xv;
  agg[i] = 0.f;
  if (f == 0) cnt[i >> 6] = 0.f;
}

// ---------------------------------------------------------------------------
// Kernel 4: edge MLP (41->256->128->64, relu,relu,linear) via WMMA,
//           scatter t into combined at src and tgt with f32 atomics.
// Block = 128 threads = 4 waves cooperating on M=64 edges:
//   - activations staged once in block-shared LDS (A/H1/H2),
//   - N-tiles split across waves, so each weight fragment is fetched from L2
//     once per 64 edges and feeds 4 WMMAs (one per 16-row tile).
// ---------------------------------------------------------------------------
__global__ __launch_bounds__(128) void k_edge_mlp(
    const float* __restrict__ nbr, const long long* __restrict__ eidx,
    const _Float16* __restrict__ W1t, const _Float16* __restrict__ W2t,
    const _Float16* __restrict__ W3t,
    const float* __restrict__ b1, const float* __restrict__ b2,
    const float* __restrict__ b3,
    float* __restrict__ combined, int E) {
  __shared__ _Float16 smem[64 * 64 + 64 * 256 + 64 * 128];  // 57344 B
  _Float16* A  = smem;                 // 64 x 64
  _Float16* H1 = A + 64 * 64;          // 64 x 256
  _Float16* H2 = H1 + 64 * 256;        // 64 x 128
  const int wave = threadIdx.x >> 5;
  const int lane = threadIdx.x & 31;
  const int eBase = blockIdx.x * 64;
  const int col = lane & 15;
  const int h   = lane >> 4;

  // Stage 64 edge rows into LDS as f16, zero-padded K=41 -> 64
  for (int idx = threadIdx.x; idx < 64 * 64; idx += 128) {
    int r = idx >> 6, k = idx & 63;
    int e = eBase + r;
    float v = (k < 41 && e < E) ? nbr[(long long)e * 41 + k] : 0.f;
    A[idx] = (_Float16)v;
  }
  __syncthreads();

  // Layer 1: (64x64)x(64x256) + b1, relu -> H1.  16 N-tiles, 4 per wave.
  for (int t = 0; t < 4; ++t) {
    int nt = wave * 4 + t;
    v8f acc[4] = {vzero8(), vzero8(), vzero8(), vzero8()};
#pragma unroll
    for (int kc = 0; kc < 2; ++kc) {
      v16h b = ldB_frag(W1t + (nt * 16 + col) * 64 + kc * 32 + h * 16);
#pragma unroll
      for (int mt = 0; mt < 4; ++mt)
        acc[mt] = wmma_f16(ldA_frag(A + mt * 16 * 64, 64, lane, kc * 32), b, acc[mt]);
    }
    float bias = b1[nt * 16 + col];
#pragma unroll
    for (int mt = 0; mt < 4; ++mt)
#pragma unroll
      for (int r = 0; r < 8; ++r)
        H1[(mt * 16 + h * 8 + r) * 256 + nt * 16 + col] =
            (_Float16)fmaxf(acc[mt][r] + bias, 0.f);
  }
  __syncthreads();

  // Layer 2: (64x256)x(256x128) + b2, relu -> H2.  8 N-tiles, 2 per wave.
  for (int t = 0; t < 2; ++t) {
    int nt = wave * 2 + t;
    v8f acc[4] = {vzero8(), vzero8(), vzero8(), vzero8()};
#pragma unroll
    for (int kc = 0; kc < 8; ++kc) {
      v16h b = ldB_frag(W2t + (nt * 16 + col) * 256 + kc * 32 + h * 16);
#pragma unroll
      for (int mt = 0; mt < 4; ++mt)
        acc[mt] = wmma_f16(ldA_frag(H1 + mt * 16 * 256, 256, lane, kc * 32), b, acc[mt]);
    }
    float bias = b2[nt * 16 + col];
#pragma unroll
    for (int mt = 0; mt < 4; ++mt)
#pragma unroll
      for (int r = 0; r < 8; ++r)
        H2[(mt * 16 + h * 8 + r) * 128 + nt * 16 + col] =
            (_Float16)fmaxf(acc[mt][r] + bias, 0.f);
  }
  __syncthreads();

  // Layer 3: (64x128)x(128x64) + b3 -> t; 4 N-tiles, 1 per wave.
  // Scatter-add t into combined[src] and combined[tgt].
  {
    int nt = wave;
    v8f acc[4] = {vzero8(), vzero8(), vzero8(), vzero8()};
#pragma unroll
    for (int kc = 0; kc < 4; ++kc) {
      v16h b = ldB_frag(W3t + (nt * 16 + col) * 128 + kc * 32 + h * 16);
#pragma unroll
      for (int mt = 0; mt < 4; ++mt)
        acc[mt] = wmma_f16(ldA_frag(H2 + mt * 16 * 128, 128, lane, kc * 32), b, acc[mt]);
    }
    float bias = b3[nt * 16 + col];
#pragma unroll
    for (int mt = 0; mt < 4; ++mt) {
      int rowbase = mt * 16 + h * 8;
#pragma unroll
      for (int r = 0; r < 8; ++r) {
        int e = eBase + rowbase + r;
        if (e < E) {
          float v = acc[mt][r] + bias;
          long long si = eidx[e], ti = eidx[E + e];
          atomicAdd(&combined[si * 64 + nt * 16 + col], v);
          atomicAdd(&combined[ti * 64 + nt * 16 + col], v);
        }
      }
    }
  }
}

// ---------------------------------------------------------------------------
// Kernel 5: EdgeConv: m = relu([x_i, x_j - x_i] @ eW1 + eb1) @ eW2 + eb2,
//           scatter-add into agg[tgt], count degrees.
// Same block-cooperative M=64 scheme as k_edge_mlp.
// ---------------------------------------------------------------------------
__global__ __launch_bounds__(128) void k_edge_conv(
    const float* __restrict__ x, const float* __restrict__ combined,
    const long long* __restrict__ eidx,
    const _Float16* __restrict__ eW1t, const _Float16* __restrict__ eW2t,
    const float* __restrict__ eb1, const float* __restrict__ eb2,
    float* __restrict__ agg, float* __restrict__ cnt, int E) {
  __shared__ _Float16 smem[64 * 128 + 64 * 256];  // 49152 B
  _Float16* A = smem;                  // 64 x 128
  _Float16* H = A + 64 * 128;          // 64 x 256
  const int wave = threadIdx.x >> 5;
  const int lane = threadIdx.x & 31;
  const int eBase = blockIdx.x * 64;
  const int col = lane & 15;
  const int h   = lane >> 4;

  // Build A rows: [x_i | x_j - x_i]  (gathers hit L2: x/combined ~51MB << 192MB)
  for (int idx = threadIdx.x; idx < 64 * 64; idx += 128) {
    int r = idx >> 6, cfe = idx & 63;
    int e = eBase + r;
    if (e < E) {
      long long si = eidx[e], ti = eidx[E + e];
      float xi = x[ti * 64 + cfe];
      float xj = combined[si * 64 + cfe];
      A[r * 128 + cfe]      = (_Float16)xi;
      A[r * 128 + 64 + cfe] = (_Float16)(xj - xi);
    } else {
      A[r * 128 + cfe]      = (_Float16)0.f;
      A[r * 128 + 64 + cfe] = (_Float16)0.f;
    }
  }
  if (threadIdx.x < 64 && (eBase + threadIdx.x) < E)
    atomicAdd(&cnt[eidx[E + eBase + threadIdx.x]], 1.f);
  __syncthreads();

  // Layer 1: (64x128)x(128x256) + eb1, relu -> H.  16 N-tiles, 4 per wave.
  for (int t = 0; t < 4; ++t) {
    int nt = wave * 4 + t;
    v8f acc[4] = {vzero8(), vzero8(), vzero8(), vzero8()};
#pragma unroll
    for (int kc = 0; kc < 4; ++kc) {
      v16h b = ldB_frag(eW1t + (nt * 16 + col) * 128 + kc * 32 + h * 16);
#pragma unroll
      for (int mt = 0; mt < 4; ++mt)
        acc[mt] = wmma_f16(ldA_frag(A + mt * 16 * 128, 128, lane, kc * 32), b, acc[mt]);
    }
    float bias = eb1[nt * 16 + col];
#pragma unroll
    for (int mt = 0; mt < 4; ++mt)
#pragma unroll
      for (int r = 0; r < 8; ++r)
        H[(mt * 16 + h * 8 + r) * 256 + nt * 16 + col] =
            (_Float16)fmaxf(acc[mt][r] + bias, 0.f);
  }
  __syncthreads();

  // Layer 2: (64x256)x(256x64) + eb2 -> m; 4 N-tiles, 1 per wave.
  // Scatter into agg[tgt].
  {
    int nt = wave;
    v8f acc[4] = {vzero8(), vzero8(), vzero8(), vzero8()};
#pragma unroll
    for (int kc = 0; kc < 8; ++kc) {
      v16h b = ldB_frag(eW2t + (nt * 16 + col) * 256 + kc * 32 + h * 16);
#pragma unroll
      for (int mt = 0; mt < 4; ++mt)
        acc[mt] = wmma_f16(ldA_frag(H + mt * 16 * 256, 256, lane, kc * 32), b, acc[mt]);
    }
    float bias = eb2[nt * 16 + col];
#pragma unroll
    for (int mt = 0; mt < 4; ++mt) {
      int rowbase = mt * 16 + h * 8;
#pragma unroll
      for (int r = 0; r < 8; ++r) {
        int e = eBase + rowbase + r;
        if (e < E)
          atomicAdd(&agg[eidx[E + e] * 64 + nt * 16 + col], acc[mt][r] + bias);
      }
    }
  }
}

// ---------------------------------------------------------------------------
// Kernel 7: out = BN(agg / max(cnt,1)) + combined
// ---------------------------------------------------------------------------
__global__ void k_final(const float* __restrict__ agg, const float* __restrict__ cnt,
                        const float* __restrict__ combined, const float* __restrict__ stats,
                        const float* __restrict__ g, const float* __restrict__ b,
                        float* __restrict__ out, int n) {
  int i = blockIdx.x * 256 + threadIdx.x;
  if (i >= n * 64) return;
  int f = i & 63;
  float inv_n = 1.f / (float)n;
  float mu  = stats[f] * inv_n;
  float var = stats[64 + f] * inv_n - mu * mu;
  float v = agg[i] / fmaxf(cnt[i >> 6], 1.f);
  out[i] = (v - mu) * rsqrtf(var + 1e-5f) * g[f] + b[f] + combined[i];
}

// ---------------------------------------------------------------------------
extern "C" void kernel_launch(void* const* d_in, const int* in_sizes, int n_in,
                              void* d_out, int out_size, void* d_ws, size_t ws_size,
                              hipStream_t stream) {
  const float*     atom  = (const float*)d_in[0];
  const float*     nbr   = (const float*)d_in[1];
  const long long* eidx  = (const long long*)d_in[2];   // int64, [2][E]
  const float* W1  = (const float*)d_in[3];
  const float* b1  = (const float*)d_in[4];
  const float* W2  = (const float*)d_in[5];
  const float* b2  = (const float*)d_in[6];
  const float* W3  = (const float*)d_in[7];
  const float* b3  = (const float*)d_in[8];
  const float* eW1 = (const float*)d_in[9];
  const float* eb1 = (const float*)d_in[10];
  const float* eW2 = (const float*)d_in[11];
  const float* eb2 = (const float*)d_in[12];
  const float* bn_in_g  = (const float*)d_in[13];
  const float* bn_in_b  = (const float*)d_in[14];
  const float* bn_out_g = (const float*)d_in[15];
  const float* bn_out_b = (const float*)d_in[16];
  float* out = (float*)d_out;

  const int N = in_sizes[0] / 64;
  const int E = in_sizes[1] / 41;

  // Workspace layout (all offsets 256B-aligned where it matters)
  char* w = (char*)d_ws;
  _Float16* W1t  = (_Float16*)(w + 0);        // 32768
  _Float16* W2t  = (_Float16*)(w + 32768);    // 65536
  _Float16* W3t  = (_Float16*)(w + 98304);    // 16384
  _Float16* eW1t = (_Float16*)(w + 114688);   // 65536
  _Float16* eW2t = (_Float16*)(w + 180224);   // 32768
  float* stats_in  = (float*)(w + 212992);    // 512
  float* stats_out = (float*)(w + 213504);    // 512
  float* xbuf      = (float*)(w + 214016);
  float* combined  = xbuf + (size_t)N * 64;
  float* agg       = combined + (size_t)N * 64;
  float* cnt       = agg + (size_t)N * 64;

  const int nElem = N * 64;
  const int gElem = (nElem + 255) / 256;
  const int gEdge = (E + 63) / 64;            // 4 waves cooperate on 64 edges

  k_prep<<<(106752 + 255) / 256, 256, 0, stream>>>(W1, W2, W3, eW1, eW2,
      W1t, W2t, W3t, eW1t, eW2t, stats_in, stats_out);
  k_stats<<<512, 256, 0, stream>>>(atom, nullptr, stats_in, N);
  k_apply_in<<<gElem, 256, 0, stream>>>(atom, stats_in, bn_in_g, bn_in_b,
      xbuf, combined, agg, cnt, N);
  k_edge_mlp<<<gEdge, 128, 0, stream>>>(nbr, eidx, W1t, W2t, W3t, b1, b2, b3,
      combined, E);
  k_edge_conv<<<gEdge, 128, 0, stream>>>(xbuf, combined, eidx, eW1t, eW2t,
      eb1, eb2, agg, cnt, E);
  k_stats<<<512, 256, 0, stream>>>(agg, cnt, stats_out, N);
  k_final<<<gElem, 256, 0, stream>>>(agg, cnt, combined, stats_out,
      bn_out_g, bn_out_b, out, N);
}
